// SSelfAttention_21474836480531
// MI455X (gfx1250) — matrix operand order repaired and checked by reading, not compiled, not run on hardware
//
#include <hip/hip_runtime.h>
#include <hip/hip_bf16.h>

#define HEADS 8
#define EMBED 128
#define HDIM  16
#define NNODE 256
#define TLEN  128

typedef __attribute__((ext_vector_type(16))) _Float16 v16h;
typedef __attribute__((ext_vector_type(8)))  float    v8f;

union F16Frag { v16h v; unsigned u[8]; };

__device__ __forceinline__ v8f wmma_f16(const F16Frag& a, const F16Frag& b, v8f c) {
    // D = A(16x32 f16) * B(32x16 f16) + C(16x16 f32)
    return __builtin_amdgcn_wmma_f32_16x16x32_f16(
        false, a.v, false, b.v, (short)0, c, false, false);
}

// ---------------------------------------------------------------------------
// Kernel 1: fused per-(t,h) projection + cross-node attention + softmax + P*V
// One block per (t,h); 256 threads = 8 wave32s. Writes O into Obuf (=d_out)
// with layout [q][t][h*16+d] so the output projection can run row-blocked.
// ---------------------------------------------------------------------------
__global__ __launch_bounds__(256) void attn_th_kernel(
    const float* __restrict__ values, const float* __restrict__ keys,
    const float* __restrict__ query,  const int*   __restrict__ mask,
    const float* __restrict__ Wv, const float* __restrict__ Wk,
    const float* __restrict__ Wq, float* __restrict__ Obuf)
{
    __shared__ __align__(16) _Float16 QpL[NNODE][32];   // K padded 16->32 with zeros
    __shared__ __align__(16) _Float16 KpL[NNODE][32];   // K padded 16->32 with zeros
    __shared__ __align__(16) _Float16 VpT[HDIM][NNODE]; // transposed: [d][node]
    __shared__ __align__(16) _Float16 PSc[8][16][32];   // per-wave P scratch
    __shared__ float WqL[256], WkL[256], WvL[256];

    const int t = blockIdx.x >> 3;
    const int h = blockIdx.x & 7;
    const int n = threadIdx.x;                          // node id, 0..255

    // stage projection weights
    WqL[n] = Wq[n]; WkL[n] = Wk[n]; WvL[n] = Wv[n];

    // single-pass read of this block's raw q/k/v slice (16 floats each)
    const size_t base = ((size_t)n * TLEN + t) * EMBED + h * HDIM;
    float qr[16], kr[16], vr[16];
    const float4* q4 = (const float4*)(query  + base);
    const float4* k4 = (const float4*)(keys   + base);
    const float4* v4 = (const float4*)(values + base);
#pragma unroll
    for (int i = 0; i < 4; ++i) {
        float4 a = q4[i]; qr[4*i+0]=a.x; qr[4*i+1]=a.y; qr[4*i+2]=a.z; qr[4*i+3]=a.w;
        float4 b = k4[i]; kr[4*i+0]=b.x; kr[4*i+1]=b.y; kr[4*i+2]=b.z; kr[4*i+3]=b.w;
        float4 c = v4[i]; vr[4*i+0]=c.x; vr[4*i+1]=c.y; vr[4*i+2]=c.z; vr[4*i+3]=c.w;
    }
    __syncthreads();

    // per-head linear projections (x @ W.T), write f16 into LDS
#pragma unroll
    for (int e = 0; e < 16; ++e) {
        float sq = 0.f, sk = 0.f, sv = 0.f;
#pragma unroll
        for (int d = 0; d < 16; ++d) {
            sq += qr[d] * WqL[e*16 + d];
            sk += kr[d] * WkL[e*16 + d];
            sv += vr[d] * WvL[e*16 + d];
        }
        QpL[n][e] = (_Float16)sq; QpL[n][16 + e] = (_Float16)0.f;
        KpL[n][e] = (_Float16)sk; KpL[n][16 + e] = (_Float16)0.f;
        VpT[e][n] = (_Float16)sv;
    }
    __syncthreads();

    const int w    = threadIdx.x >> 5;
    const int lane = threadIdx.x & 31;
    const int l16  = lane & 15;
    const int kbA  = (lane >> 4) * 16;   // A-fragment K base (contiguous layout)
    const int kbB  = (lane >> 4) * 16;   // B-fragment K base
    const float SCALE = 0.08838834764831845f;  // 1/sqrt(128)

#pragma unroll 1
    for (int qq = 0; qq < 2; ++qq) {
        const int qt = w + 8 * qq;       // q tile 0..15
        // A fragment of projected Q (invariant over key chunks)
        const int Mq = qt * 16 + l16;
        F16Frag qa;
#pragma unroll
        for (int j = 0; j < 4; ++j) {
            qa.u[j]     = *(const unsigned*)&QpL[Mq][kbA + 2*j];
            qa.u[4 + j] = *(const unsigned*)&QpL[Mq][kbA + 8 + 2*j]; // padded zeros for lanes 16..31
        }

        v8f acc = {0.f,0.f,0.f,0.f,0.f,0.f,0.f,0.f};
        float rs[8] = {0.f,0.f,0.f,0.f,0.f,0.f,0.f,0.f};

#pragma unroll 1
        for (int kc = 0; kc < 8; ++kc) {   // 32 key-nodes per chunk
            // B fragments of Kp^T for the two 16-wide key tiles
            const int n0 = (2*kc) * 16 + l16;
            const int n1 = n0 + 16;
            F16Frag b0, b1;
#pragma unroll
            for (int j = 0; j < 8; ++j) {
                b0.u[j] = *(const unsigned*)&KpL[n0][kbB + 2*j];
                b1.u[j] = *(const unsigned*)&KpL[n1][kbB + 2*j];
            }
            v8f zc = {0.f,0.f,0.f,0.f,0.f,0.f,0.f,0.f};
            v8f e0 = wmma_f16(qa, b0, zc);
            v8f e1 = wmma_f16(qa, b1, zc);

            // mask -> scale -> clip -> exp; stash P (f16) for the A fragment
            const int col0 = 2*kc*16 + l16;
            const int col1 = col0 + 16;
#pragma unroll
            for (int r = 0; r < 8; ++r) {
                const int Mr = (lane < 16) ? r : r + 8;
                const int q  = qt * 16 + Mr;
                const int m0 = mask[q * NNODE + col0];
                const int m1 = mask[q * NNODE + col1];
                float s0 = (m0 == 0) ? 5.0f : fminf(fmaxf(e0[r] * SCALE, -5.0f), 5.0f);
                float s1 = (m1 == 0) ? 5.0f : fminf(fmaxf(e1[r] * SCALE, -5.0f), 5.0f);
                float p0 = __expf(s0);
                float p1 = __expf(s1);
                rs[r] += p0 + p1;
                PSc[w][Mr][l16]      = (_Float16)p0;
                PSc[w][Mr][16 + l16] = (_Float16)p1;
            }
            // per-wave LDS is in-order; wait for the P stores before re-reading
            asm volatile("s_wait_dscnt 0" ::: "memory");

            // A fragment of P (16 q-rows x 32 k-cols)
            F16Frag pa;
#pragma unroll
            for (int j = 0; j < 4; ++j) {
                pa.u[j]     = *(const unsigned*)&PSc[w][l16][kbA + 2*j];
                pa.u[4 + j] = *(const unsigned*)&PSc[w][l16][kbA + 8 + 2*j];
            }
            // B fragment of V chunk (32 k-rows x 16 d-cols) from transposed store
            F16Frag vb;
#pragma unroll
            for (int j = 0; j < 8; ++j)
                vb.u[j] = *(const unsigned*)&VpT[l16][kc*32 + kbB + 2*j];

            acc = wmma_f16(pa, vb, acc);
        }

        // softmax denominator: reduce over the 16 lanes holding each row
#pragma unroll
        for (int r = 0; r < 8; ++r) {
            float s = rs[r];
            s += __shfl_xor(s, 1, 32);
            s += __shfl_xor(s, 2, 32);
            s += __shfl_xor(s, 4, 32);
            s += __shfl_xor(s, 8, 32);
            rs[r] = s;
        }
        // normalize and store O (coalesced 64B per half-wave per row)
#pragma unroll
        for (int r = 0; r < 8; ++r) {
            const int Mr = (lane < 16) ? r : r + 8;
            const int q  = qt * 16 + Mr;
            float o = acc[r] / rs[r];
            Obuf[((size_t)q * TLEN + t) * EMBED + h * HDIM + l16] = o;
        }
    }
}

// ---------------------------------------------------------------------------
// Kernel 2: in-place output projection  X = X @ Wout^T + bout
// Row-blocked (64 rows/block): each block stages its rows in LDS first, so
// rewriting the same global rows after __syncthreads() is safe.
// ---------------------------------------------------------------------------
__global__ __launch_bounds__(256) void out_proj_kernel(
    float* __restrict__ X, const float* __restrict__ Wout,
    const float* __restrict__ bout)
{
    __shared__ __align__(16) _Float16 XL[64][128];    // 16 KB
    __shared__ __align__(16) _Float16 WoL[128][128];  // 32 KB, row-major [n][k]
    __shared__ float boutL[128];

    const int tid = threadIdx.x;
    const size_t rowbase = (size_t)blockIdx.x * 64;

#pragma unroll
    for (int i = 0; i < 64; ++i) {           // 16384 weights / 256 threads
        int idx = i * 256 + tid;
        ((_Float16*)WoL)[idx] = (_Float16)Wout[idx];
    }
#pragma unroll
    for (int i = 0; i < 32; ++i) {           // 8192 elements / 256 threads
        int idx = i * 256 + tid;
        int r = idx >> 7, c = idx & 127;
        XL[r][c] = (_Float16)X[(rowbase + r) * EMBED + c];
    }
    if (tid < 128) boutL[tid] = bout[tid];
    __syncthreads();

    const int w    = tid >> 5;
    const int lane = tid & 31;
    const int l16  = lane & 15;
    const int kbA  = (lane >> 4) * 16;
    const int kbB  = (lane >> 4) * 16;
    const int tileRow = w >> 1;
    const int Ma = tileRow * 16 + l16;

#pragma unroll 1
    for (int tc = 0; tc < 4; ++tc) {
        const int tileCol = (w & 1) * 4 + tc;
        const int nb = tileCol * 16 + l16;
        v8f acc = {0.f,0.f,0.f,0.f,0.f,0.f,0.f,0.f};
#pragma unroll
        for (int c = 0; c < 4; ++c) {        // K = 128 in 4 chunks of 32
            F16Frag a, b;
#pragma unroll
            for (int j = 0; j < 4; ++j) {
                a.u[j]     = *(const unsigned*)&XL[Ma][c*32 + kbA + 2*j];
                a.u[4 + j] = *(const unsigned*)&XL[Ma][c*32 + kbA + 8 + 2*j];
            }
#pragma unroll
            for (int j = 0; j < 8; ++j)
                b.u[j] = *(const unsigned*)&WoL[nb][c*32 + kbB + 2*j];
            acc = wmma_f16(a, b, acc);
        }
        const float bb = boutL[nb];
#pragma unroll
        for (int r = 0; r < 8; ++r) {
            int row = (int)rowbase + tileRow * 16 + ((lane < 16) ? r : r + 8);
            X[(size_t)row * EMBED + nb] = acc[r] + bb;
        }
    }
}

extern "C" void kernel_launch(void* const* d_in, const int* in_sizes, int n_in,
                              void* d_out, int out_size, void* d_ws, size_t ws_size,
                              hipStream_t stream) {
    (void)in_sizes; (void)n_in; (void)d_ws; (void)ws_size; (void)out_size;
    const float* values = (const float*)d_in[0];
    const float* keys   = (const float*)d_in[1];
    const float* query  = (const float*)d_in[2];
    const int*   mask   = (const int*)d_in[3];
    const float* Wv     = (const float*)d_in[4];
    const float* Wk     = (const float*)d_in[5];
    const float* Wq     = (const float*)d_in[6];
    const float* Wout   = (const float*)d_in[7];
    const float* bout   = (const float*)d_in[8];
    float* out = (float*)d_out;

    attn_th_kernel<<<dim3(TLEN * HEADS), dim3(256), 0, stream>>>(
        values, keys, query, mask, Wv, Wk, Wq, out);
    out_proj_kernel<<<dim3((NNODE * TLEN) / 64), dim3(256), 0, stream>>>(
        out, Wout, bout);
}
